// MultiHeadAttention_89120571392303
// MI455X (gfx1250) — compile-verified, matrix operand order checked by reading, not run on hardware
//
#include <hip/hip_runtime.h>
#include <hip/hip_bf16.h>

// MI455X / gfx1250, wave32, WMMA bf16 + Tensor Data Mover pipeline.
// pack(f32->bf16) -> fused QKV GEMM (TDM double-buffered, WMMA) ->
// flash attention (TDM double-buffered K/V, WMMA, online softmax) ->
// fused output-proj + LayerNorm (TDM, WMMA).

typedef __bf16 bf16;
typedef __attribute__((ext_vector_type(16))) __bf16 v16bf;
typedef __attribute__((ext_vector_type(8)))  float  v8f;
typedef unsigned int u32x4 __attribute__((ext_vector_type(4)));
typedef int          i32x4 __attribute__((ext_vector_type(4)));
typedef int          i32x8 __attribute__((ext_vector_type(8)));

#define H_    12
#define S_    1024
#define D_    768
#define DH_   64
#define B_    8
#define MTOT  (B_ * S_)    // 8192 rows
#define NQKV  (3 * D_)     // 2304 fused QKV columns
#define LN_EPS 1e-5f

// ---------------------------------------------------------------------------
// helpers
// ---------------------------------------------------------------------------
__device__ __forceinline__ bf16 f2bf(float f) {
  unsigned u = __builtin_bit_cast(unsigned int, f);
  unsigned r = u + 0x7fffu + ((u >> 16) & 1u);      // round-to-nearest-even
  unsigned short h = (unsigned short)(r >> 16);
  return __builtin_bit_cast(bf16, h);
}

__device__ __forceinline__ v8f wmma_bf16(v16bf a, v16bf b, v8f c) {
  return __builtin_amdgcn_wmma_f32_16x16x32_bf16(
      /*neg_a=*/false, a, /*neg_b=*/false, b,
      /*c_mod=*/(short)0, c, /*reuse_a=*/false, /*reuse_b=*/false);
}

// A(16x32)/B(32x16) fragment where K is the contiguous axis; outer strides ldo.
// Lanes 0-15: outer=0..15, K={0..7,16..23}; lanes 16-31: same outer, K={8..15,24..31}.
__device__ __forceinline__ v16bf ld_frag_kmaj(const bf16* p, int ldo, int lane) {
  int half = (lane >> 4) & 1;
  int o    = lane & 15;
  const bf16* q = p + o * ldo + 8 * half;
  v16bf r;
#pragma unroll
  for (int e = 0; e < 16; ++e) {
    int k = (e & 7) + ((e >> 3) << 4);   // e + 8*(e>=8)
    r[e] = q[k];
  }
  return r;
}

// Fragment where the OUTER (M/N) axis is contiguous and K strides by ldk
// (used for B-frags gathered from a row-major [K][N] tile).
__device__ __forceinline__ v16bf ld_frag_kstride(const bf16* p, int ldk, int lane) {
  int half = (lane >> 4) & 1;
  int o    = lane & 15;
  const bf16* q = p + o + 8 * half * ldk;
  v16bf r;
#pragma unroll
  for (int e = 0; e < 16; ++e) {
    int k = (e & 7) + ((e >> 3) << 4);
    r[e] = q[k * ldk];
  }
  return r;
}

// ---------------------------------------------------------------------------
// Tensor Data Mover: 2D tile load global -> LDS (ISA cdna5 ch.8 D# layout).
// data_size = 1 byte. tensor_dim0 = stride_bytes (row allocation), tensor_dim1
// = nrows, tile = row_bytes x nrows, rows packed contiguously in LDS.
// Issue from one wave only; completion via s_wait_tensorcnt + barrier.
// ---------------------------------------------------------------------------
__device__ __forceinline__ void tdm_load_2d(const void* lds_dst, const void* gsrc,
                                            unsigned row_bytes, unsigned nrows,
                                            unsigned stride_bytes) {
  unsigned lds = (unsigned)(unsigned long long)lds_dst;  // LDS aperture: addr[31:0]
  unsigned long long ga = (unsigned long long)gsrc;
  u32x4 g0;
  g0[0] = 1u;                                   // valid descriptor count = 1
  g0[1] = lds;                                  // lds_addr (bytes)
  g0[2] = (unsigned)ga;                         // global_addr[31:0]
  g0[3] = (unsigned)(ga >> 32) | (2u << 30);    // global_addr[56:32] | type=2
  i32x8 g1;
  g1[0] = 0;                                    // no multicast, data_size=1B
  g1[1] = (int)((stride_bytes & 0xffffu) << 16);                    // tensor_dim0[15:0]
  g1[2] = (int)((stride_bytes >> 16) | ((nrows & 0xffffu) << 16));  // td0[31:16], td1[15:0]
  g1[3] = (int)((nrows >> 16) | (row_bytes << 16));                 // td1[31:16], tile_dim0
  g1[4] = (int)(nrows & 0xffffu);               // tile_dim1 (tile_dim2 = 0)
  g1[5] = (int)stride_bytes;                    // tensor_dim0_stride[31:0]
  g1[6] = 0;                                    // stride[47:32], td1_stride lo
  g1[7] = 0;
  i32x4 z4 = (i32x4)0;
  i32x8 z8 = (i32x8)0;
  __builtin_amdgcn_tensor_load_to_lds(g0, g1, z4, z4, z8, 0);
}

// ---------------------------------------------------------------------------
// pack / convert kernels (one-time, tiny vs GEMM cost)
// ---------------------------------------------------------------------------
__global__ void k_f32_to_bf16(const float* __restrict__ src,
                              bf16* __restrict__ dst, int n) {
  int i = (blockIdx.x * blockDim.x + threadIdx.x) * 4;
  if (i + 3 < n) {
    float4 v = *(const float4*)(src + i);
    dst[i + 0] = f2bf(v.x); dst[i + 1] = f2bf(v.y);
    dst[i + 2] = f2bf(v.z); dst[i + 3] = f2bf(v.w);
  }
}

// WqkvT[n][d], n = which*768 + h*64 + e  (transposed for contiguous B rows)
__global__ void k_pack_wqkvT(const float* __restrict__ Wq,
                             const float* __restrict__ Wk,
                             const float* __restrict__ Wv,
                             bf16* __restrict__ WT) {
  int id = blockIdx.x * 256 + threadIdx.x;   // id = n*768 + d
  if (id >= NQKV * D_) return;
  int n = id / D_, d = id % D_;
  int which = n / D_, hn = n % D_;
  int h = hn >> 6, e = hn & 63;
  const float* W = (which == 0) ? Wq : (which == 1) ? Wk : Wv;
  WT[id] = f2bf(W[(h * D_ + d) * DH_ + e]);
}

__global__ void k_pack_bias(const float* __restrict__ bq,
                            const float* __restrict__ bk,
                            const float* __restrict__ bv,
                            float* __restrict__ bqkv) {
  int n = blockIdx.x * 256 + threadIdx.x;
  if (n >= NQKV) return;
  int which = n / D_, hn = n % D_;
  const float* b = (which == 0) ? bq : (which == 1) ? bk : bv;
  bqkv[n] = b[hn];
}

// WoT[n][k] = Wo[k][n]  (Wo is [H*DH=768, D=768] row-major K x N)
__global__ void k_pack_woT(const float* __restrict__ Wo, bf16* __restrict__ WoT) {
  int id = blockIdx.x * 256 + threadIdx.x;   // id = n*768 + k
  if (id >= D_ * D_) return;
  int n = id / D_, k = id % D_;
  WoT[id] = f2bf(Wo[k * D_ + n]);
}

// ---------------------------------------------------------------------------
// Fused QKV projection: X[8192,768] @ WqkvT^T -> Q/K/V bf16 [B,H,S,64]
// 256 threads / 8 waves, tile 128x128, BK=32. TDM double-buffered tiles:
// one barrier per K-step; wave0 DMAs the next tiles while WMMAs run.
// ---------------------------------------------------------------------------
__global__ __launch_bounds__(256) void k_qkv_gemm(
    const bf16* __restrict__ X, const bf16* __restrict__ WT,
    const float* __restrict__ bqkv,
    bf16* __restrict__ Q, bf16* __restrict__ K, bf16* __restrict__ V) {
  __shared__ __align__(16) bf16 As[2][128][32];
  __shared__ __align__(16) bf16 BT[2][128][32];

  int t = threadIdx.x, lane = t & 31, w = t >> 5;
  int wm = w >> 1, wn = w & 1;                // 4x2 wave grid -> 32x64 per wave
  int bm = blockIdx.y, bn = blockIdx.x;
  int idx = lane & 15, half = lane >> 4;

  const bf16* xrow = X  + (size_t)(bm * 128) * D_;
  const bf16* wrow = WT + (size_t)(bn * 128) * D_;

  v8f acc[2][4];
#pragma unroll
  for (int i = 0; i < 2; ++i)
#pragma unroll
    for (int j = 0; j < 4; ++j) acc[i][j] = (v8f)(0.0f);

  if (w == 0) {                               // prologue: DMA first tiles
    tdm_load_2d(&As[0][0][0], xrow, 64, 128, D_ * 2);
    tdm_load_2d(&BT[0][0][0], wrow, 64, 128, D_ * 2);
  }

  for (int kt = 0; kt < D_ / 32; ++kt) {
    int cur = kt & 1;
    if (w == 0) __builtin_amdgcn_s_wait_tensorcnt(0);
    __syncthreads();                          // tiles[cur] ready; [1-cur] free
    if (w == 0 && kt + 1 < D_ / 32) {         // overlap next DMA with compute
      tdm_load_2d(&As[1 - cur][0][0], xrow + (kt + 1) * 32, 64, 128, D_ * 2);
      tdm_load_2d(&BT[1 - cur][0][0], wrow + (kt + 1) * 32, 64, 128, D_ * 2);
    }
#pragma unroll
    for (int mf = 0; mf < 2; ++mf) {
      v16bf a = ld_frag_kmaj(&As[cur][wm * 32 + mf * 16][0], 32, lane);
#pragma unroll
      for (int nf = 0; nf < 4; ++nf) {
        v16bf b = ld_frag_kmaj(&BT[cur][wn * 64 + nf * 16][0], 32, lane);
        acc[mf][nf] = wmma_bf16(a, b, acc[mf][nf]);
      }
    }
  }

  // epilogue: bias, scale (Q only), scatter into [B,H,S,DH] bf16
#pragma unroll
  for (int mf = 0; mf < 2; ++mf)
#pragma unroll
    for (int nf = 0; nf < 4; ++nf) {
      int n = bn * 128 + wn * 64 + nf * 16 + idx;     // 0..2303
      int which = n / D_, hn = n % D_;
      int h = hn >> 6, e = hn & 63;
      float bias = bqkv[n];
      float scl = (which == 0) ? 0.125f : 1.0f;       // 1/sqrt(64)
      bf16* dst = (which == 0) ? Q : (which == 1) ? K : V;
#pragma unroll
      for (int r = 0; r < 8; ++r) {
        int m = bm * 128 + wm * 32 + mf * 16 + r + 8 * half;
        int bidx = m >> 10, s = m & 1023;
        dst[(size_t)((bidx * H_ + h) * S_ + s) * DH_ + e] =
            f2bf((acc[mf][nf][r] + bias) * scl);
      }
    }
}

// ---------------------------------------------------------------------------
// Flash attention: block = (b, h, 128 query rows); each wave owns 16 rows.
// K/V tiles (64 keys, contiguous 8KB each) DMA'd by TDM, double-buffered;
// one barrier per key block; P staged per-wave in LDS (s_wait_dscnt only).
// ---------------------------------------------------------------------------
__global__ __launch_bounds__(256) void k_flash_attn(
    const bf16* __restrict__ Q, const bf16* __restrict__ K,
    const bf16* __restrict__ V, const unsigned char* __restrict__ mask,
    bf16* __restrict__ ctx) {
  __shared__ __align__(16) bf16 Ks[2][64][64];  // [key][dh]
  __shared__ __align__(16) bf16 Vs[2][64][64];  // [key][dh]
  __shared__ __align__(16) bf16 Ps[8][16][64];  // per-wave P staging

  int t = threadIdx.x, lane = t & 31, w = t >> 5;
  int idx = lane & 15, half = lane >> 4;
  int bh = blockIdx.y, b = bh / H_, h = bh % H_;
  int qrow0 = blockIdx.x * 128 + w * 16;

  const bf16* qp = Q + (size_t)(bh * S_ + qrow0) * DH_;
  const bf16* kp = K + (size_t)bh * S_ * DH_;
  const bf16* vp = V + (size_t)bh * S_ * DH_;

  v16bf qa0 = ld_frag_kmaj(qp, DH_, lane);        // K-dim 0..31
  v16bf qa1 = ld_frag_kmaj(qp + 32, DH_, lane);   // K-dim 32..63

  v8f o[4];
#pragma unroll
  for (int i = 0; i < 4; ++i) o[i] = (v8f)(0.0f);
  float rmax[8], rsum[8];
#pragma unroll
  for (int r = 0; r < 8; ++r) { rmax[r] = -3.0e38f; rsum[r] = 0.0f; }

  if (w == 0) {                               // prologue DMA: first key block
    tdm_load_2d(&Ks[0][0][0], kp, 8192, 1, 8192);
    tdm_load_2d(&Vs[0][0][0], vp, 8192, 1, 8192);
  }

  for (int jb = 0; jb < S_ / 64; ++jb) {
    int cur = jb & 1;
    int j0 = jb * 64;
    if (w == 0) __builtin_amdgcn_s_wait_tensorcnt(0);
    __syncthreads();                          // bufs[cur] ready; [1-cur] free
    if (w == 0 && jb + 1 < S_ / 64) {         // DMA next block during compute
      tdm_load_2d(&Ks[1 - cur][0][0], kp + (size_t)(j0 + 64) * DH_, 8192, 1, 8192);
      tdm_load_2d(&Vs[1 - cur][0][0], vp + (size_t)(j0 + 64) * DH_, 8192, 1, 8192);
    }

    // scores S = Q K^T (1/sqrt(dh) folded into Q) + key-padding mask
    v8f sf[4];
#pragma unroll
    for (int nf = 0; nf < 4; ++nf) {
      v16bf b0 = ld_frag_kmaj(&Ks[cur][nf * 16][0], 64, lane);
      v16bf b1 = ld_frag_kmaj(&Ks[cur][nf * 16][32], 64, lane);
      v8f z = (v8f)(0.0f);
      sf[nf] = wmma_bf16(qa0, b0, z);
      sf[nf] = wmma_bf16(qa1, b1, sf[nf]);
      float madd = mask[b * S_ + j0 + nf * 16 + idx] ? 0.0f : -3.0e38f;
#pragma unroll
      for (int r = 0; r < 8; ++r) sf[nf][r] += madd;
    }

    // online softmax: row max/sum via width-16 shuffles (C-layout halves)
    float mnew[8], cf[8];
#pragma unroll
    for (int r = 0; r < 8; ++r) {
      float mx = fmaxf(fmaxf(sf[0][r], sf[1][r]), fmaxf(sf[2][r], sf[3][r]));
#pragma unroll
      for (int d = 1; d < 16; d <<= 1) mx = fmaxf(mx, __shfl_xor(mx, d, 16));
      mnew[r] = fmaxf(rmax[r], mx);
      cf[r] = __expf(rmax[r] - mnew[r]);
      rmax[r] = mnew[r];
    }
    float psum[8];
#pragma unroll
    for (int r = 0; r < 8; ++r) psum[r] = 0.0f;
#pragma unroll
    for (int nf = 0; nf < 4; ++nf)
#pragma unroll
      for (int r = 0; r < 8; ++r) {
        float p = __expf(sf[nf][r] - mnew[r]);
        psum[r] += p;
        Ps[w][r + 8 * half][nf * 16 + idx] = f2bf(p);
      }
#pragma unroll
    for (int r = 0; r < 8; ++r) {
#pragma unroll
      for (int d = 1; d < 16; d <<= 1) psum[r] += __shfl_xor(psum[r], d, 16);
      rsum[r] = rsum[r] * cf[r] + psum[r];
#pragma unroll
      for (int nf = 0; nf < 4; ++nf) o[nf][r] *= cf[r];
    }

    // Ps is wave-private: only our own LDS stores must land (DScnt==0)
    asm volatile("s_wait_dscnt 0x0" ::: "memory");

    v16bf pa0 = ld_frag_kmaj(&Ps[w][0][0], 64, lane);
    v16bf pa1 = ld_frag_kmaj(&Ps[w][0][32], 64, lane);
#pragma unroll
    for (int nf = 0; nf < 4; ++nf) {          // O += P V   (B from [key][dh])
      v16bf b0 = ld_frag_kstride(&Vs[cur][0][nf * 16], 64, lane);
      v16bf b1 = ld_frag_kstride(&Vs[cur][32][nf * 16], 64, lane);
      o[nf] = wmma_bf16(pa0, b0, o[nf]);
      o[nf] = wmma_bf16(pa1, b1, o[nf]);
    }
  }

  // normalize and write ctx bf16 [B,S, h*64+dh]
#pragma unroll
  for (int nf = 0; nf < 4; ++nf)
#pragma unroll
    for (int r = 0; r < 8; ++r) {
      int row = qrow0 + r + 8 * half;
      int col = h * DH_ + nf * 16 + idx;
      ctx[(size_t)(b * S_ + row) * D_ + col] = f2bf(o[nf][r] / rsum[r]);
    }
}

// ---------------------------------------------------------------------------
// Output projection + LayerNorm fused: ctx[8192,768] @ Wo + bo, then LN.
// Block computes 32 full rows (N=768 resident) -> LN via LDS row stats.
// TDM loads both panels (B panel = 48KB, single-buffered).
// ---------------------------------------------------------------------------
__global__ __launch_bounds__(256) void k_proj_ln(
    const bf16* __restrict__ Actx, const bf16* __restrict__ WoT,
    const float* __restrict__ bo, const float* __restrict__ gamma,
    const float* __restrict__ beta, float* __restrict__ out) {
  __shared__ __align__(16) bf16 As[32][32];
  __shared__ __align__(16) bf16 BT[768][32];
  __shared__ float rowsum[32], rowsq[32], mu_s[32], rstd_s[32];

  int t = threadIdx.x, lane = t & 31, w = t >> 5;
  int idx = lane & 15, half = lane >> 4;
  int mf = w >> 2, nq = w & 3;
  int bm = blockIdx.x;
  const bf16* arow = Actx + (size_t)(bm * 32) * D_;

  if (t < 32) { rowsum[t] = 0.0f; rowsq[t] = 0.0f; }

  v8f acc[12];
#pragma unroll
  for (int f = 0; f < 12; ++f) acc[f] = (v8f)(0.0f);

  for (int kk = 0; kk < D_; kk += 32) {
    if (w == 0) {
      tdm_load_2d(&As[0][0], arow + kk, 64, 32, D_ * 2);
      tdm_load_2d(&BT[0][0], WoT + kk, 64, 768, D_ * 2);
      __builtin_amdgcn_s_wait_tensorcnt(0);
    }
    __syncthreads();
    v16bf a = ld_frag_kmaj(&As[mf * 16][0], 32, lane);
#pragma unroll
    for (int f = 0; f < 12; ++f) {
      v16bf bfr = ld_frag_kmaj(&BT[nq * 192 + f * 16][0], 32, lane);
      acc[f] = wmma_bf16(a, bfr, acc[f]);
    }
    __syncthreads();
  }

  // bias + per-row partial stats
  float ps[8], pq[8];
#pragma unroll
  for (int r = 0; r < 8; ++r) { ps[r] = 0.0f; pq[r] = 0.0f; }
#pragma unroll
  for (int f = 0; f < 12; ++f) {
    int col = nq * 192 + f * 16 + idx;
    float bb = bo[col];
#pragma unroll
    for (int r = 0; r < 8; ++r) {
      float v2 = acc[f][r] + bb;
      acc[f][r] = v2;
      ps[r] += v2; pq[r] += v2 * v2;
    }
  }
#pragma unroll
  for (int r = 0; r < 8; ++r) {
#pragma unroll
    for (int d = 1; d < 16; d <<= 1) {
      ps[r] += __shfl_xor(ps[r], d, 16);
      pq[r] += __shfl_xor(pq[r], d, 16);
    }
  }
  if (idx == 0) {
#pragma unroll
    for (int r = 0; r < 8; ++r) {
      int row = mf * 16 + r + 8 * half;
      atomicAdd(&rowsum[row], ps[r]);           // ds_add_f32
      atomicAdd(&rowsq[row], pq[r]);
    }
  }
  __syncthreads();
  if (t < 32) {
    float mu = rowsum[t] * (1.0f / 768.0f);
    float var = rowsq[t] * (1.0f / 768.0f) - mu * mu;
    mu_s[t] = mu;
    rstd_s[t] = rsqrtf(var + LN_EPS);
  }
  __syncthreads();
#pragma unroll
  for (int f = 0; f < 12; ++f) {
    int col = nq * 192 + f * 16 + idx;
    float g2 = gamma[col], bt = beta[col];
#pragma unroll
    for (int r = 0; r < 8; ++r) {
      int row = mf * 16 + r + 8 * half;
      out[(size_t)(bm * 32 + row) * D_ + col] =
          (acc[f][r] - mu_s[row]) * rstd_s[row] * g2 + bt;
    }
  }
}

// ---------------------------------------------------------------------------
// host launcher
// ---------------------------------------------------------------------------
extern "C" void kernel_launch(void* const* d_in, const int* in_sizes, int n_in,
                              void* d_out, int out_size, void* d_ws, size_t ws_size,
                              hipStream_t stream) {
  const float*         x    = (const float*)d_in[0];
  const unsigned char* mask = (const unsigned char*)d_in[1];  // jax bool -> 1B
  const float* Wq = (const float*)d_in[2];
  const float* bq = (const float*)d_in[3];
  const float* Wk = (const float*)d_in[4];
  const float* bk = (const float*)d_in[5];
  const float* Wv = (const float*)d_in[6];
  const float* bv = (const float*)d_in[7];
  const float* Wo = (const float*)d_in[8];
  const float* bo = (const float*)d_in[9];
  const float* gamma = (const float*)d_in[10];
  const float* beta  = (const float*)d_in[11];
  float* out = (float*)d_out;

  char* ws = (char*)d_ws;
  size_t off = 0;
  bf16*  Xbf   = (bf16*)(ws + off);  off += (size_t)MTOT * D_ * 2;        // 12.6 MB
  bf16*  WqkvT = (bf16*)(ws + off);  off += (size_t)NQKV * D_ * 2;        //  3.5 MB
  float* bqkv  = (float*)(ws + off); off += (size_t)NQKV * 4;
  bf16*  WoT   = (bf16*)(ws + off);  off += (size_t)D_ * D_ * 2;          //  1.2 MB
  bf16*  Qb    = (bf16*)(ws + off);  off += (size_t)B_ * H_ * S_ * DH_ * 2;
  bf16*  Kb    = (bf16*)(ws + off);  off += (size_t)B_ * H_ * S_ * DH_ * 2;
  bf16*  Vb    = (bf16*)(ws + off);  off += (size_t)B_ * H_ * S_ * DH_ * 2;
  bf16*  Ctx   = (bf16*)(ws + off);  off += (size_t)MTOT * D_ * 2;        // ~68 MB

  (void)in_sizes; (void)n_in; (void)out_size; (void)ws_size;

  int nX = MTOT * D_;
  k_f32_to_bf16<<<nX / (256 * 4), 256, 0, stream>>>(x, Xbf, nX);
  k_pack_wqkvT<<<(NQKV * D_) / 256, 256, 0, stream>>>(Wq, Wk, Wv, WqkvT);
  k_pack_bias<<<(NQKV + 255) / 256, 256, 0, stream>>>(bq, bk, bv, bqkv);
  k_pack_woT<<<(D_ * D_) / 256, 256, 0, stream>>>(Wo, WoT);

  dim3 gQkv(NQKV / 128, MTOT / 128);            // (18, 64)
  k_qkv_gemm<<<gQkv, 256, 0, stream>>>(Xbf, WqkvT, bqkv, Qb, Kb, Vb);

  dim3 gAtt(S_ / 128, B_ * H_);                 // (8, 96)
  k_flash_attn<<<gAtt, 256, 0, stream>>>(Qb, Kb, Vb, mask, Ctx);

  k_proj_ln<<<MTOT / 32, 256, 0, stream>>>(Ctx, WoT, bo, gamma, beta, out);
}
